// Vq3dNet_72962904425031
// MI455X (gfx1250) — matrix operand order repaired and checked by reading, not compiled, not run on hardware
//
#include <hip/hip_runtime.h>
#include <hip/hip_bf16.h>

// ---------------------------------------------------------------------------
// VQ-VAE 3D vector quantization for MI455X (gfx1250, wave32, WMMA bf16).
//
// Shapes: feat [2,128,48,48,48] f32, embedding [512,128] f32,
//         ema_cluster_size [512] f32, ema_w [512,128] f32.
// N = 2*48^3 = 221184 voxels, D = 128, K = 512.
// ---------------------------------------------------------------------------

#define KCODES   512
#define DDIM     128
#define V3       110592          // 48^3
#define NVOX     221184          // 2 * 48^3
#define TILE_N   128             // voxels per workgroup (divides V3)
#define CHUNK2   32              // codes per double-buffered LDS chunk
#define NCHUNK   (KCODES / CHUNK2)
#define LDA      136             // padded row stride (bf16 elems): 272B, 16B-aligned
#define DECAY    0.99f
#define EPS_     1e-5f
#define COMMIT   0.25f

// d_out layout (floats), tuple order of the reference:
#define OFF_LOSS   0
#define OFF_QUANT  1
#define OFF_PERP   (1 + 28311552)                 // 28311553
#define OFF_NEWEMB (OFF_PERP + 1)                 // 28311554
#define OFF_CS     (OFF_NEWEMB + 65536)           // 28377090
#define OFF_EMAW   (OFF_CS + 512)                 // 28377602
#define OFF_IDX    (OFF_EMAW + 65536)             // 28443138

// workspace layout (bytes):
#define WS_EMBBF   0          // 512*128*2 = 131072  (bf16 codebook)
#define WS_ENORM   131072     // 512*4
#define WS_COUNTS  133120     // 512*4
#define WS_DW      135168     // 512*128*4 = 262144
#define WS_LOSS    397312     // 4
#define WS_IDX     397376     // 221184*4 = 884736  -> total ~1.28 MB

typedef __bf16 bf16_t;
typedef __attribute__((ext_vector_type(16))) __bf16 v16bf;
typedef __attribute__((ext_vector_type(8)))  float  v8f;
typedef __attribute__((ext_vector_type(4)))  unsigned int u32x4;

// GCC-style int4 vector matching the async-LDS builtin's parameter type.
typedef int v4i_g __attribute__((vector_size(16)));
typedef __attribute__((address_space(1))) v4i_g* gv4p;   // global int4*
typedef __attribute__((address_space(3))) v4i_g* lv4p;   // LDS int4*

union FragBF { v16bf v; u32x4 q[2]; };

#if __has_builtin(__builtin_amdgcn_global_load_async_to_lds_b128)
#define HAVE_ASYNC_LDS 1
#else
#define HAVE_ASYNC_LDS 0
#endif

__device__ __forceinline__ void wait_async_lds() {
#if HAVE_ASYNC_LDS
#if __has_builtin(__builtin_amdgcn_s_wait_asynccnt)
    __builtin_amdgcn_s_wait_asynccnt(0);
#else
    asm volatile("s_wait_asynccnt 0x0" ::: "memory");
#endif
#endif
}

__device__ __forceinline__ unsigned short f2bf(float x) {
    unsigned int u = __float_as_uint(x);
    unsigned int r = u + 0x7FFFu + ((u >> 16) & 1u);   // round-to-nearest-even
    return (unsigned short)(r >> 16);
}

// ---------------------------------------------------------------- prep 1 ----
__global__ __launch_bounds__(256)
void vq_prep1(const float* __restrict__ emb, unsigned short* __restrict__ embBf,
              float* __restrict__ dw) {
    int i = blockIdx.x * 256 + threadIdx.x;          // 65536 elements
    if (i < KCODES * DDIM) {
        embBf[i] = f2bf(emb[i]);
        dw[i] = 0.0f;
    }
}

// ---------------------------------------------------------------- prep 2 ----
__global__ __launch_bounds__(256)
void vq_prep2(const float* __restrict__ emb, float* __restrict__ eNorm,
              float* __restrict__ counts, float* __restrict__ lossAcc) {
    int k = blockIdx.x * 256 + threadIdx.x;
    if (k < KCODES) {
        float s = 0.0f;
        const float* row = emb + (size_t)k * DDIM;
        #pragma unroll 8
        for (int d = 0; d < DDIM; ++d) { float e = row[d]; s += e * e; }
        eNorm[k] = s;
        counts[k] = 0.0f;
        if (k == 0) lossAcc[0] = 0.0f;
    }
}

// --------------------------------------------- distances + argmin (WMMA) ----
__global__ __launch_bounds__(256)
void vq_dist_argmin(const float* __restrict__ feat,
                    const unsigned short* __restrict__ embBf,
                    const float* __restrict__ eNorm,
                    int* __restrict__ idxInt,
                    float* __restrict__ idxOutF) {
    __shared__ __align__(16) unsigned short As[TILE_N * LDA];       // 34816 B
    __shared__ __align__(16) unsigned short Bs[2][CHUNK2 * LDA];    // 17408 B
    __shared__ float NsAll[KCODES];                                 //  2048 B

    const int tid = threadIdx.x;
    const int voxbase = blockIdx.x * TILE_N;
    const int b  = voxbase / V3;
    const int v0 = voxbase % V3;

    // Async (ASYNCcnt-tracked) or fallback copy of one 32-code bf16 chunk.
    auto load_chunk = [&](int buf, int cbase) {
#if HAVE_ASYNC_LDS
        #pragma unroll
        for (int r = 0; r < 2; ++r) {
            int e = (r * 256 + tid) * 16;            // byte offset within chunk
            int c = e >> 8, inner = e & 255;         // code row, byte within row
            gv4p src = (gv4p)((const char*)embBf + (size_t)cbase * 256 + e);
            lv4p dst = (lv4p)((char*)&Bs[buf][0] + c * (LDA * 2) + inner);
            __builtin_amdgcn_global_load_async_to_lds_b128(src, dst, 0, 0);
        }
#else
        for (int i = tid; i < CHUNK2 * DDIM / 4; i += 256) {
            int c = i >> 5, d4 = (i & 31) * 4;
            *(uint2*)&Bs[buf][c * LDA + d4] =
                *(const uint2*)&embBf[(size_t)(cbase + c) * DDIM + d4];
        }
#endif
    };

    // Stage A tile: feat[b, d, v0..v0+127] -> As[v][d] as bf16 (coalesced per d).
    for (int i = tid; i < TILE_N * DDIM; i += 256) {
        int d = i >> 7, v = i & 127;
        float x = feat[((size_t)(b * DDIM + d)) * V3 + v0 + v];
        As[v * LDA + d] = f2bf(x);
    }
    for (int i = tid; i < KCODES; i += 256) NsAll[i] = eNorm[i];
    load_chunk(0, 0);                                // prefetch chunk 0
    wait_async_lds();
    __syncthreads();

    const int wav  = tid >> 5;
    const int lane = tid & 31;
    const int lm   = lane & 15;
    const int half = lane >> 4;

    // Cache this wave's 4 A-fragments (M-tile = 16 voxels, K-chunks of 32).
    const unsigned short* arow = &As[(wav * 16 + lm) * LDA];
    FragBF af[4];
    #pragma unroll
    for (int j = 0; j < 4; ++j) {
        int kc = 32 * j;
        af[j].q[0] = *(const u32x4*)(arow + kc + half * 16);       // K 0-7 / 16-23
        af[j].q[1] = *(const u32x4*)(arow + kc + half * 16 + 8);   // K 8-15 / 24-31
    }

    float bestVal[8];
    int   bestIdx[8];
    #pragma unroll
    for (int v = 0; v < 8; ++v) { bestVal[v] = 3.4e38f; bestIdx[v] = 0; }

    for (int ci = 0; ci < NCHUNK; ++ci) {
        const int cur = ci & 1;
        if (ci + 1 < NCHUNK) load_chunk(cur ^ 1, (ci + 1) * CHUNK2);  // prefetch
        const int cbase = ci * CHUNK2;

        const unsigned short* b0row = &Bs[cur][lm * LDA];
        const unsigned short* b1row = &Bs[cur][(16 + lm) * LDA];
        float bias0 = NsAll[cbase + lm];
        float bias1 = NsAll[cbase + 16 + lm];

        v8f acc0, acc1;
        #pragma unroll
        for (int v = 0; v < 8; ++v) { acc0[v] = 0.0f; acc1[v] = 0.0f; }

        // Two independent accumulation chains -> WMMA/DS-load overlap.
        #pragma unroll
        for (int j = 0; j < 4; ++j) {
            int off = 32 * j + half * 16;            // K 0-15 (lanes<16) / 16-31
            FragBF b0, b1;
            b0.q[0] = *(const u32x4*)(b0row + off);
            b0.q[1] = *(const u32x4*)(b0row + off + 8);
            b1.q[0] = *(const u32x4*)(b1row + off);
            b1.q[1] = *(const u32x4*)(b1row + off + 8);
            acc0 = __builtin_amdgcn_wmma_f32_16x16x32_bf16(
                       false, af[j].v, false, b0.v, (short)0, acc0, false, false);
            acc1 = __builtin_amdgcn_wmma_f32_16x16x32_bf16(
                       false, af[j].v, false, b1.v, (short)0, acc1, false, false);
        }

        int code0 = cbase + lm;
        int code1 = cbase + 16 + lm;
        #pragma unroll
        for (int v = 0; v < 8; ++v) {
            float d0 = bias0 - 2.0f * acc0[v];       // ||x||^2 constant/row: dropped
            if (d0 < bestVal[v]) { bestVal[v] = d0; bestIdx[v] = code0; }
            float d1 = bias1 - 2.0f * acc1[v];
            if (d1 < bestVal[v]) { bestVal[v] = d1; bestIdx[v] = code1; }
        }

        wait_async_lds();        // own async issues complete
        __syncthreads();         // all waves' prefetches visible; cur reusable
    }

    // Reduce argmin across the 16 lanes holding the same voxel row.
    #pragma unroll
    for (int v = 0; v < 8; ++v) {
        float bv = bestVal[v];
        int   bi = bestIdx[v];
        #pragma unroll
        for (int off = 1; off < 16; off <<= 1) {
            float ov = __shfl_xor(bv, off, 32);
            int   oi = __shfl_xor(bi, off, 32);
            if (ov < bv || (ov == bv && oi < bi)) { bv = ov; bi = oi; }
        }
        if (lm == 0) {
            int m = v + 8 * half;                    // voxel row within wave tile
            int n = voxbase + wav * 16 + m;
            idxInt[n]  = bi;
            idxOutF[n] = (float)bi;
        }
    }
}

// --------------------------------- quantize + loss + counts/dw scatter ------
__global__ __launch_bounds__(256)
void vq_quantize(const float* __restrict__ feat,
                 const float* __restrict__ emb,
                 const int* __restrict__ idxInt,
                 float* __restrict__ qout,
                 float* __restrict__ dw,
                 float* __restrict__ counts,
                 float* __restrict__ lossAcc) {
    __shared__ int sIdx[TILE_N];
    __shared__ float sRed[8];

    const int tid = threadIdx.x;
    const int voxbase = blockIdx.x * TILE_N;
    const int b  = voxbase / V3;
    const int v0 = voxbase % V3;

    for (int i = tid; i < TILE_N; i += 256) sIdx[i] = idxInt[voxbase + i];
    __syncthreads();

    float lsum = 0.0f;
    for (int i = tid; i < TILE_N * DDIM; i += 256) {
        int d = i >> 7, v = i & 127;
        int k = sIdx[v];
        size_t gi = ((size_t)(b * DDIM + d)) * V3 + v0 + v;
        float x = feat[gi];
        float q = emb[k * DDIM + d];
        qout[gi] = q;                               // straight-through value == q
        float df = q - x;
        lsum += df * df;
        atomicAdd(&dw[k * DDIM + d], x);
    }
    for (int i = tid; i < TILE_N; i += 256) atomicAdd(&counts[sIdx[i]], 1.0f);

    // block-reduce lsum
    #pragma unroll
    for (int off = 16; off > 0; off >>= 1) lsum += __shfl_down(lsum, off, 32);
    if ((tid & 31) == 0) sRed[tid >> 5] = lsum;
    __syncthreads();
    if (tid == 0) {
        float t = 0.0f;
        #pragma unroll
        for (int w = 0; w < 8; ++w) t += sRed[w];
        atomicAdd(lossAcc, t);
    }
}

// ------------------------------------------------ finalize (EMA, stats) -----
__global__ __launch_bounds__(512)
void vq_finalize(const float* __restrict__ ema_cs,
                 const float* __restrict__ ema_w,
                 const float* __restrict__ counts,
                 const float* __restrict__ dw,
                 const float* __restrict__ lossAcc,
                 float* __restrict__ out) {
    __shared__ float red[512];
    const int k = threadIdx.x;                      // 0..511

    float cnt = counts[k];
    float cs1 = ema_cs[k] * DECAY + (1.0f - DECAY) * cnt;

    red[k] = cs1;
    __syncthreads();
    for (int s = 256; s > 0; s >>= 1) {
        if (k < s) red[k] += red[k + s];
        __syncthreads();
    }
    float n = red[0];
    __syncthreads();

    float csF = (cs1 + EPS_) / (n + KCODES * EPS_) * n;
    out[OFF_CS + k] = csF;

    float p = cnt / (float)NVOX;
    red[k] = p * logf(p + 1e-10f);
    __syncthreads();
    for (int s = 256; s > 0; s >>= 1) {
        if (k < s) red[k] += red[k + s];
        __syncthreads();
    }
    if (k == 0) {
        out[OFF_PERP] = expf(-red[0]);
        out[OFF_LOSS] = COMMIT * lossAcc[0] / ((float)NVOX * (float)DDIM);
    }

    const float* wr = ema_w + (size_t)k * DDIM;
    const float* dr = dw + (size_t)k * DDIM;
    float inv = 1.0f / csF;
    #pragma unroll 4
    for (int d = 0; d < DDIM; ++d) {
        float w = wr[d] * DECAY + (1.0f - DECAY) * dr[d];
        out[OFF_EMAW + k * DDIM + d]   = w;
        out[OFF_NEWEMB + k * DDIM + d] = w * inv;
    }
}

// ---------------------------------------------------------------------------
extern "C" void kernel_launch(void* const* d_in, const int* in_sizes, int n_in,
                              void* d_out, int out_size, void* d_ws, size_t ws_size,
                              hipStream_t stream) {
    const float* feat   = (const float*)d_in[0];
    const float* emb    = (const float*)d_in[1];
    const float* ema_cs = (const float*)d_in[2];
    const float* ema_w  = (const float*)d_in[3];
    float* out = (float*)d_out;

    unsigned char* ws = (unsigned char*)d_ws;
    unsigned short* embBf  = (unsigned short*)(ws + WS_EMBBF);
    float*          eNorm  = (float*)(ws + WS_ENORM);
    float*          counts = (float*)(ws + WS_COUNTS);
    float*          dw     = (float*)(ws + WS_DW);
    float*          lossAcc= (float*)(ws + WS_LOSS);
    int*            idxInt = (int*)(ws + WS_IDX);

    vq_prep1<<<(KCODES * DDIM + 255) / 256, 256, 0, stream>>>(emb, embBf, dw);
    vq_prep2<<<(KCODES + 255) / 256, 256, 0, stream>>>(emb, eNorm, counts, lossAcc);

    vq_dist_argmin<<<NVOX / TILE_N, 256, 0, stream>>>(
        feat, embBf, eNorm, idxInt, out + OFF_IDX);

    vq_quantize<<<NVOX / TILE_N, 256, 0, stream>>>(
        feat, emb, idxInt, out + OFF_QUANT, dw, counts, lossAcc);

    vq_finalize<<<1, 512, 0, stream>>>(ema_cs, ema_w, counts, dw, lossAcc, out);
}
